// attention_head_78134045049089
// MI455X (gfx1250) — matrix operand order repaired and checked by reading, not compiled, not run on hardware
//
#include <hip/hip_runtime.h>
#include <hip/hip_bf16.h>

typedef __attribute__((ext_vector_type(16))) _Float16 v16h;
typedef __attribute__((ext_vector_type(8)))  _Float16 v8h;
typedef __attribute__((ext_vector_type(8)))  float    v8f;

#define WMMA_F16(a, b, c) \
  __builtin_amdgcn_wmma_f32_16x16x32_f16(false, (a), false, (b), (short)0, (c), false, false)

constexpr int BB = 8, TT = 2048, CC = 1024, HH = 64;
constexpr int NROW = BB * TT;               // 16384
constexpr float SCALE = 0.125f;             // 1/sqrt(64)
constexpr float INV_BLOCK = 1.0f / 2048.0f;

// ---------------------------------------------------------------------------
// Kernel 0: convert Wq/Wk/Wv (C x H, f32, row-major) -> WT f16 [mat][h][c]
// (K-contiguous so B-fragments of the projection GEMM are contiguous loads)
// ---------------------------------------------------------------------------
__global__ void k_prep_w(const float* __restrict__ Wq, const float* __restrict__ Wk,
                         const float* __restrict__ Wv, _Float16* __restrict__ wt) {
  int idx = blockIdx.x * 256 + threadIdx.x;
  if (idx >= 3 * HH * CC) return;
  int mat = idx / (HH * CC);
  int r   = idx % (HH * CC);
  int h   = r / CC, c = r % CC;
  const float* W = (mat == 0) ? Wq : (mat == 1) ? Wk : Wv;
  wt[idx] = (_Float16)W[c * HH + h];
}

// ---------------------------------------------------------------------------
// Kernel 1: q/k/v projection via v_wmma_f32_16x16x32_f16.
// Grid = 3 mats (block-uniform -> scalar store branch) x 128 row-blocks.
// Each wave: one 16-row tile, ALL 4 h-tiles of its matrix (4 accumulators),
// so every A-fragment (x rows, f32->f16 on the fly) feeds 4 WMMAs.
//  q,k stored row-major f16 [b*T][64];  v stored transposed f16 [b][64][T].
// ---------------------------------------------------------------------------
__global__ void k_proj(const float* __restrict__ x, const _Float16* __restrict__ wt,
                       _Float16* __restrict__ qo, _Float16* __restrict__ ko,
                       _Float16* __restrict__ vT) {
  int wid  = threadIdx.x >> 5;
  int lane = threadIdx.x & 31;
  int mat     = blockIdx.x / 128;                 // block-uniform: 0=q 1=k 2=v
  int rowTile = (blockIdx.x % 128) * 8 + wid;     // 0 .. 1023 (16 rows each)
  int hi  = lane >> 4;                            // half-wave select
  int ln  = lane & 15;

  const float*    xrow  = x  + (size_t)(rowTile * 16 + ln) * CC;
  const _Float16* wbase = wt + (size_t)mat * HH * CC + (size_t)ln * CC + 16 * hi;

  v8f acc[4] = {{}, {}, {}, {}};
  for (int c0 = 0; c0 < CC; c0 += 32) {
    if (c0 + 256 < CC) __builtin_prefetch(xrow + c0 + 256, 0, 0);
    // A: x rows, f32 -> f16 on the fly (contiguous K: k = 16*hi + s)
    v8f xa = *(const v8f*)(xrow + c0 + 16 * hi);
    v8f xb = *(const v8f*)(xrow + c0 + 16 * hi + 8);
    v16h a;
#pragma unroll
    for (int i = 0; i < 8; ++i) { a[i] = (_Float16)xa[i]; a[8 + i] = (_Float16)xb[i]; }
    // B: WT[h][c], K contiguous (32B per lane); 4 h-tiles reuse one A-frag
#pragma unroll
    for (int t = 0; t < 4; ++t) {
      v16h b = *(const v16h*)(wbase + (size_t)t * 16 * CC + c0);
      acc[t] = WMMA_F16(a, b, acc[t]);
    }
  }

  // C-layout: element (m = j + 8*hi, n = ln); mat is scalar -> uniform branch
  int rbase = rowTile * 16 + 8 * hi;
  if (mat == 0) {
#pragma unroll
    for (int t = 0; t < 4; ++t)
#pragma unroll
      for (int j = 0; j < 8; ++j)
        qo[(size_t)(rbase + j) * HH + t * 16 + ln] = (_Float16)acc[t][j];
  } else if (mat == 1) {
#pragma unroll
    for (int t = 0; t < 4; ++t)
#pragma unroll
      for (int j = 0; j < 8; ++j)
        ko[(size_t)(rbase + j) * HH + t * 16 + ln] = (_Float16)acc[t][j];
  } else {
    int bb = rbase / TT;
#pragma unroll
    for (int t = 0; t < 4; ++t)
#pragma unroll
      for (int j = 0; j < 8; ++j) {
        int row = rbase + j;
        vT[((size_t)bb * HH + t * 16 + ln) * TT + (row % TT)] = (_Float16)acc[t][j];
      }
  }
}

// ---------------------------------------------------------------------------
// Kernel 2: fused causal flash attention + adaptive-span reweight.
// 8 independent waves per block; each wave owns a 16-row query tile and its
// own 16x32 LDS P-tile.  Key loop (32 keys/iter): 4 QK wmma + 4 PV wmma.
// Online state per row: running max m; Z = sum(e) and S = sum(e*span) kept as
// LANE-PARTIAL accumulators (alpha is row-uniform), reduced once at the end.
// out = acc / (S + 1e-8 * Z).
// ---------------------------------------------------------------------------
__global__ void k_attn(const _Float16* __restrict__ qf, const _Float16* __restrict__ kf,
                       const _Float16* __restrict__ vT, const float* __restrict__ cvp,
                       float* __restrict__ out) {
  __shared__ _Float16 lds[8 * 16 * 32];
  int wid  = threadIdx.x >> 5;
  int lane = threadIdx.x & 31;
  int hi   = lane >> 4;
  int ln   = lane & 15;
  _Float16* pbuf = lds + wid * (16 * 32);

  int b  = blockIdx.x / (TT / 128);
  int q0 = (blockIdx.x % (TT / 128)) * 128 + wid * 16;
  float cv = cvp[0];

  // Q A-fragments (K = head dim, 2 chunks of 32)
  const _Float16* qrow = qf + ((size_t)b * TT + q0 + ln) * HH;
  v16h a0, a1;
  {
    v8h s00 = *(const v8h*)(qrow + 16 * hi);
    v8h s01 = *(const v8h*)(qrow + 16 * hi + 8);
    v8h s10 = *(const v8h*)(qrow + 32 + 16 * hi);
    v8h s11 = *(const v8h*)(qrow + 40 + 16 * hi);
#pragma unroll
    for (int i = 0; i < 8; ++i) {
      a0[i] = s00[i]; a0[8 + i] = s01[i];
      a1[i] = s10[i]; a1[8 + i] = s11[i];
    }
  }

  float mrow[8], Zp[8], Sp[8];          // Zp/Sp: lane-partial sums
#pragma unroll
  for (int j = 0; j < 8; ++j) { mrow[j] = -1e30f; Zp[j] = 0.f; Sp[j] = 0.f; }
  v8f acc0 = {}, acc1 = {}, acc2 = {}, acc3 = {};

  const _Float16* kbase = kf + (size_t)b * TT * HH;
  const _Float16* vbase = vT + (size_t)b * HH * TT;
  int kmax = q0 + 15;

  for (int s0 = 0; s0 <= kmax; s0 += 32) {
    // --- QK^T: two 16-key subtiles, each K=64 -> 2 wmma
    v16h bk0 = *(const v16h*)(kbase + (size_t)(s0 + ln) * HH + 16 * hi);
    v16h bk1 = *(const v16h*)(kbase + (size_t)(s0 + ln) * HH + 32 + 16 * hi);
    v16h bk2 = *(const v16h*)(kbase + (size_t)(s0 + 16 + ln) * HH + 16 * hi);
    v16h bk3 = *(const v16h*)(kbase + (size_t)(s0 + 16 + ln) * HH + 32 + 16 * hi);
    v8f sc0 = {}, sc1 = {};
    sc0 = WMMA_F16(a0, bk0, sc0);
    sc0 = WMMA_F16(a1, bk1, sc0);
    sc1 = WMMA_F16(a0, bk2, sc1);
    sc1 = WMMA_F16(a1, bk3, sc1);

    int kc0 = s0 + ln, kc1 = s0 + 16 + ln;
    float sp0 = fminf(fmaxf((float)(kc0 + 1) * INV_BLOCK + cv, 0.f), 1.f);
    float sp1 = fminf(fmaxf((float)(kc1 + 1) * INV_BLOCK + cv, 0.f), 1.f);

#pragma unroll
    for (int j = 0; j < 8; ++j) {
      int qr = q0 + j + 8 * hi;
      float x0 = (kc0 <= qr) ? sc0[j] * SCALE : -1e30f;
      float x1 = (kc1 <= qr) ? sc1[j] * SCALE : -1e30f;
      float tm = fmaxf(x0, x1);
#pragma unroll
      for (int off = 8; off >= 1; off >>= 1) tm = fmaxf(tm, __shfl_xor(tm, off, 32));
      float nm    = fmaxf(mrow[j], tm);
      float alpha = __expf(mrow[j] - nm);        // row-uniform across the half-wave
      float e0 = (kc0 <= qr) ? __expf(x0 - nm) : 0.f;
      float e1 = (kc1 <= qr) ? __expf(x1 - nm) : 0.f;
      float pw0 = e0 * sp0, pw1 = e1 * sp1;
      Zp[j] = Zp[j] * alpha + (e0 + e1);         // lane-partial, no per-tile reduce
      Sp[j] = Sp[j] * alpha + (pw0 + pw1);
      mrow[j] = nm;
      acc0[j] *= alpha; acc1[j] *= alpha; acc2[j] *= alpha; acc3[j] *= alpha;
      // scatter span-weighted probs into the wave-local P tile (row m, 32 keys)
      int m = j + 8 * hi;
      pbuf[m * 32 + ln]      = (_Float16)pw0;
      pbuf[m * 32 + 16 + ln] = (_Float16)pw1;
    }

    // wave-local LDS RAW: DS ops are in-order per wave; wait + compiler fence
    asm volatile("s_wait_dscnt 0" ::: "memory");

    // P as A-fragment (16x32, K = key index)
    v16h pa;
    {
      v8h p0 = *(const v8h*)(pbuf + ln * 32 + 16 * hi);
      v8h p1 = *(const v8h*)(pbuf + ln * 32 + 16 * hi + 8);
#pragma unroll
      for (int i = 0; i < 8; ++i) { pa[i] = p0[i]; pa[8 + i] = p1[i]; }
    }
    asm volatile("" ::: "memory");

    // --- PV: V^T rows are K-contiguous -> 4 B-fragments, 4 wmma
    v16h bv0 = *(const v16h*)(vbase + (size_t)(ln)      * TT + s0 + 16 * hi);
    v16h bv1 = *(const v16h*)(vbase + (size_t)(16 + ln) * TT + s0 + 16 * hi);
    v16h bv2 = *(const v16h*)(vbase + (size_t)(32 + ln) * TT + s0 + 16 * hi);
    v16h bv3 = *(const v16h*)(vbase + (size_t)(48 + ln) * TT + s0 + 16 * hi);
    acc0 = WMMA_F16(pa, bv0, acc0);
    acc1 = WMMA_F16(pa, bv1, acc1);
    acc2 = WMMA_F16(pa, bv2, acc2);
    acc3 = WMMA_F16(pa, bv3, acc3);
  }

  // epilogue: reduce lane-partial Z/S across the 16 lanes of each half-wave,
  // then out = acc / (S + 1e-8 * Z)
#pragma unroll
  for (int j = 0; j < 8; ++j) {
    float sZ = Zp[j], sS = Sp[j];
#pragma unroll
    for (int off = 8; off >= 1; off >>= 1) {
      sZ += __shfl_xor(sZ, off, 32);
      sS += __shfl_xor(sS, off, 32);
    }
    float inv = 1.0f / (sS + 1e-8f * sZ);
    int row = q0 + j + 8 * hi;
    float* orow = out + ((size_t)b * TT + row) * HH;
    orow[ln]      = acc0[j] * inv;
    orow[16 + ln] = acc1[j] * inv;
    orow[32 + ln] = acc2[j] * inv;
    orow[48 + ln] = acc3[j] * inv;
  }
}

// ---------------------------------------------------------------------------
extern "C" void kernel_launch(void* const* d_in, const int* in_sizes, int n_in,
                              void* d_out, int out_size, void* d_ws, size_t ws_size,
                              hipStream_t stream) {
  const float* x  = (const float*)d_in[0];
  const float* Wq = (const float*)d_in[1];
  const float* Wk = (const float*)d_in[2];
  const float* Wv = (const float*)d_in[3];
  const float* cv = (const float*)d_in[4];
  float* out = (float*)d_out;

  // workspace layout (f16 units): WT | q | k | vT   (~6.4 MB total)
  _Float16* ws = (_Float16*)d_ws;
  _Float16* wt = ws;
  _Float16* qf = wt + (size_t)3 * HH * CC;
  _Float16* kf = qf + (size_t)NROW * HH;
  _Float16* vT = kf + (size_t)NROW * HH;

  // 1) weight transpose + f16 convert: 3*64*1024 elements
  k_prep_w<<<(3 * HH * CC + 255) / 256, 256, 0, stream>>>(Wq, Wk, Wv, wt);

  // 2) projections: 3 mats x 128 row-blocks; each wave = 16 rows x 64 cols
  k_proj<<<3 * 128, 256, 0, stream>>>(x, wt, qf, kf, vT);

  // 3) fused flash attention + adaptive span: 8 waves x 16 query rows / block
  k_attn<<<BB * (TT / 128), 256, 0, stream>>>(qf, kf, vT, cv, out);
}
